// PromptDPM_69922067579068
// MI455X (gfx1250) — matrix operand order repaired and checked by reading, not compiled, not run on hardware
//
#include <hip/hip_runtime.h>
#include <hip/hip_bf16.h>

// ---------------------------------------------------------------------------
// EGNN eps-net for MI455X (gfx1250). Edge MLPs (the ~170 GFLOP hot path) run
// as v_wmma_f32_16x16x32_f16 with M = 16-edge tiles per wave.
//  - A operands staged in LDS, loaded as 2x ds_load_b128 per fragment
//  - B operands pre-swizzled in workspace to the ISA fragment layout:
//    one contiguous 32B load per lane per fragment
//  - 8 N-tile accumulators live per wave (independent back-to-back WMMAs)
// ---------------------------------------------------------------------------

typedef __attribute__((ext_vector_type(16))) _Float16 v16h;
typedef __attribute__((ext_vector_type(8)))  _Float16 v8h;
typedef __attribute__((ext_vector_type(8)))  float    v8f;

#define NCC   14
#define HIDD  128
#define KRAD  224   // 196 padded to multiple of 32
#define KE1   416   // 128+128+128+32
#define KE2   128

__device__ __forceinline__ float fast_silu(float v) {
    return v * __builtin_amdgcn_rcpf(1.0f + __expf(-v));
}

// ---------------------------------------------------------------------------
// Per-wave shared staging for the edge pipeline
// ---------------------------------------------------------------------------
struct WaveSmem {
    float    cd[16 * NCC * 3];   // 2688 B
    float    sumsq[16];
    float    cscale[16];
    int      rowi[16];
    int      coli[16];
    int      etyp[16];
    float    out[16 * HIDD];     // 8192 B : final m (f32)
    _Float16 A1[16 * KRAD];      // 7168 B : radial A, then m1 A (16x128)
    _Float16 A2[16 * KE1];       // 13312 B: e1 A
};                               // 31680 B per wave, 2 waves/block

// ---------------------------------------------------------------------------
// 16x128 GEMM: A (16xK f16, LDS) * B (pre-swizzled f16, global) + bias, silu.
// A layout (ISA 16-bit A 16x32): lane m = lane&15; fragment halves are
//   [base+0..7, base+16..23], base = ks*32 + (lane&16 ? 8 : 0)   (16B aligned)
// B pre-swizzled: halves[((ks*8+nt)*32 + lane)*16 + p]
// C/D: VGPR r -> M = r + (lane>=16?8:0), N = lane&15
// ---------------------------------------------------------------------------
template <bool F16OUT>
__device__ __forceinline__ void wave_gemm_silu(const _Float16* __restrict__ Alds,
                                               int Kpad, int ksteps,
                                               const _Float16* __restrict__ Bsw,
                                               const float* __restrict__ bias,
                                               float* __restrict__ outf32,
                                               _Float16* __restrict__ outf16,
                                               int ostride, int ocol, int lane)
{
    const int m    = lane & 15;
    const int m0   = (lane & 16) ? 8 : 0;
    const int ncol = lane & 15;
    const int aoff = (lane & 16) ? 8 : 0;

    v8f acc[8];
#pragma unroll
    for (int nt = 0; nt < 8; ++nt)
#pragma unroll
        for (int r = 0; r < 8; ++r) acc[nt][r] = 0.0f;

    for (int ks = 0; ks < ksteps; ++ks) {
        const _Float16* ap = Alds + m * Kpad + ks * 32 + aoff;
        const v8h alo = *(const v8h*)ap;          // ds_load_b128
        const v8h ahi = *(const v8h*)(ap + 16);   // ds_load_b128
        const v16h a = __builtin_shufflevector(alo, ahi,
                0, 1, 2, 3, 4, 5, 6, 7, 8, 9, 10, 11, 12, 13, 14, 15);
        const _Float16* bp = Bsw + ks * 4096 + lane * 16;
#pragma unroll
        for (int nt = 0; nt < 8; ++nt) {
            const v16h b = *(const v16h*)(bp + nt * 512);   // 32B contiguous
            acc[nt] = __builtin_amdgcn_wmma_f32_16x16x32_f16(
                false, a, false, b, (short)0, acc[nt], false, false);
        }
    }

#pragma unroll
    for (int nt = 0; nt < 8; ++nt) {
        const int n = nt * 16 + ncol;
        const float bv = bias[n];
#pragma unroll
        for (int r = 0; r < 8; ++r) {
            const float v = fast_silu(acc[nt][r] + bv);
            if (F16OUT)
                outf16[(m0 + r) * ostride + ocol + n] = (_Float16)v;
            else
                outf32[(m0 + r) * HIDD + n] = v;
        }
    }
}

// ---------------------------------------------------------------------------
// Edge pipeline: one wave = 16 edges; block = 2 waves.
// ---------------------------------------------------------------------------
__global__ void __launch_bounds__(64)
edge_kernel(const float* __restrict__ xcur, const float* __restrict__ att,
            const float* __restrict__ h,
            const int* __restrict__ ctx_e, int Ectx,
            const int* __restrict__ int_e, int Eint,
            const float* __restrict__ edge_table,
            const _Float16* __restrict__ Wr,  const float* __restrict__ rb,
            const _Float16* __restrict__ We1, const float* __restrict__ e1b,
            const _Float16* __restrict__ We2, const float* __restrict__ e2b,
            const float* __restrict__ xw, const float* __restrict__ xb,
            float* __restrict__ agg, float* __restrict__ xacc)
{
    __shared__ WaveSmem sm[2];
    const int wave = threadIdx.x >> 5;
    const int lane = threadIdx.x & 31;
    WaveSmem& s = sm[wave];
    const int E     = Ectx + Eint;
    const int ebase = blockIdx.x * 32 + wave * 16;

    // 1. edge indices + edge type (concat order: ctx then inter)
    if (lane < 16) {
        int ge = ebase + lane;
        if (ge >= E) ge = E - 1;
        int r, c, t;
        if (ge < Ectx) { r = ctx_e[ge]; c = ctx_e[Ectx + ge]; t = 0; }
        else { const int g = ge - Ectx; r = int_e[g]; c = int_e[Eint + g]; t = 1; }
        s.rowi[lane] = r; s.coli[lane] = c; s.etyp[lane] = t;
        s.sumsq[lane] = 0.0f;
    }
    __syncthreads();

    // 2. coordinate differences
    for (int idx = lane; idx < 16 * NCC * 3; idx += 32) {
        const int e = idx / (NCC * 3), j = idx % (NCC * 3);
        s.cd[idx] = xcur[s.rowi[e] * (NCC * 3) + j] - xcur[s.coli[e] * (NCC * 3) + j];
    }
    __syncthreads();

    // 3. Frobenius-norm accumulation of attention-weighted Gram matrix
    for (int idx = lane; idx < 16 * NCC * NCC; idx += 32) {
        const int e  = idx / (NCC * NCC), k = idx % (NCC * NCC);
        const int ci = k / NCC, cj = k % NCC;
        const float* a = &s.cd[e * (NCC * 3) + ci * 3];
        const float* b = &s.cd[e * (NCC * 3) + cj * 3];
        const float dot = a[0] * b[0] + a[1] * b[1] + a[2] * b[2];
        const float w = att[s.rowi[e] * NCC + ci] * att[s.coli[e] * NCC + cj];
        const float v = dot * w;
        atomicAdd(&s.sumsq[e], v * v);
    }
    __syncthreads();

    // 4a. build radial A matrix (16 x 224, f16, zero-padded)
    for (int idx = lane; idx < 16 * KRAD; idx += 32) {
        const int e = idx / KRAD, k = idx % KRAD;
        float v = 0.0f;
        if (k < NCC * NCC) {
            const int ci = k / NCC, cj = k % NCC;
            const float* a = &s.cd[e * (NCC * 3) + ci * 3];
            const float* b = &s.cd[e * (NCC * 3) + cj * 3];
            const float dot = a[0] * b[0] + a[1] * b[1] + a[2] * b[2];
            const float w = att[s.rowi[e] * NCC + ci] * att[s.coli[e] * NCC + cj];
            v = dot * w / (1.0f + sqrtf(s.sumsq[e]));
        }
        s.A1[idx] = (_Float16)v;
    }
    // 4b. fill e1 A columns that don't depend on rf: [h[row] | h[col] | .. | edge_attr]
    for (int idx = lane; idx < 16 * KE1; idx += 32) {
        const int e = idx / KE1, k = idx % KE1;
        if (k >= 256 && k < 384) continue;      // rf columns written by GEMM1
        float v;
        if (k < 128)       v = h[s.rowi[e] * HIDD + k];
        else if (k < 256)  v = h[s.coli[e] * HIDD + (k - 128)];
        else               v = edge_table[s.etyp[e] * 32 + (k - 384)];
        s.A2[idx] = (_Float16)v;
    }
    __syncthreads();

    // 5. rf = silu(radial @ Wr + rb) -> written f16 straight into e1 A cols 256..383
    wave_gemm_silu<true>(s.A1, KRAD, KRAD / 32, Wr, rb, nullptr, s.A2, KE1, 256, lane);
    __syncthreads();

    // 6. m1 = silu(e1A @ We1 + e1b) -> written f16 straight into A1 (16x128)
    wave_gemm_silu<true>(s.A2, KE1, KE1 / 32, We1, e1b, nullptr, s.A1, KE2, 0, lane);
    __syncthreads();

    // 7. m = silu(m1 @ We2 + e2b) -> f32 (needed for cscale + scatter)
    wave_gemm_silu<false>(s.A1, KE2, KE2 / 32, We2, e2b, s.out, nullptr, 0, 0, lane);
    __syncthreads();

    // 8. coordinate scale per edge: cscale = m @ xw + xb
    if (lane < 16) {
        float acc = xb[0];
        for (int f = 0; f < HIDD; ++f) acc += s.out[lane * HIDD + f] * xw[f];
        s.cscale[lane] = acc;
    }
    __syncthreads();

    // 9. scatter-add: agg[row] += m ; xacc[row] += cd * cscale
    for (int idx = lane; idx < 16 * HIDD; idx += 32) {
        const int e = idx >> 7;
        if (ebase + e < E)
            atomicAdd(&agg[s.rowi[e] * HIDD + (idx & 127)], s.out[idx]);
    }
    for (int idx = lane; idx < 16 * NCC * 3; idx += 32) {
        const int e = idx / (NCC * 3);
        if (ebase + e < E)
            atomicAdd(&xacc[s.rowi[e] * (NCC * 3) + (idx % (NCC * 3))],
                      s.cd[idx] * s.cscale[e]);
    }
}

// ---------------------------------------------------------------------------
// helpers
// ---------------------------------------------------------------------------
__global__ void zero_kernel(float* p, int n) {
    const int i = blockIdx.x * blockDim.x + threadIdx.x;
    if (i < n) p[i] = 0.0f;
}

// Pre-swizzle weight matrix (Kx128 f32) into WMMA B-fragment order (f16):
// dst[((ks*8+nt)*32 + lane)*16 + p] = W[(ks*32 + (lane&16?8:0) + (p<8?p:p+8))*128 + nt*16 + (lane&15)]
__global__ void swizzle_b_kernel(const float* __restrict__ src, _Float16* __restrict__ dst,
                                 int Ksrc, int ksteps) {
    const int idx = blockIdx.x * blockDim.x + threadIdx.x;
    if (idx >= ksteps * 4096) return;
    const int p    = idx & 15;
    const int lane = (idx >> 4) & 31;
    const int frag = idx >> 9;
    const int nt   = frag & 7;
    const int ks   = frag >> 3;
    const int k = ks * 32 + ((lane & 16) ? 8 : 0) + (p < 8 ? p : p + 8);
    const int n = nt * 16 + (lane & 15);
    dst[idx] = (_Float16)((k < Ksrc) ? src[k * HIDD + n] : 0.0f);
}

__global__ void centers_accum_kernel(const float* __restrict__ X,
                                     const unsigned char* __restrict__ mgen,
                                     const unsigned char* __restrict__ amask,
                                     const int* __restrict__ bid,
                                     float* __restrict__ csum, float* __restrict__ ccnt, int N) {
    const int i = blockIdx.x * blockDim.x + threadIdx.x;
    if (i >= N) return;
    if (!mgen[i] && amask[i * NCC + 1]) {
        const int b = bid[i];
        atomicAdd(&csum[b * 3 + 0], X[i * NCC * 3 + 3 + 0]);   // channel 1 (CA)
        atomicAdd(&csum[b * 3 + 1], X[i * NCC * 3 + 3 + 1]);
        atomicAdd(&csum[b * 3 + 2], X[i * NCC * 3 + 3 + 2]);
        atomicAdd(&ccnt[b], 1.0f);
    }
}

__global__ void deg_accum_kernel(const int* __restrict__ ctx_e, int Ectx,
                                 const int* __restrict__ int_e, int Eint,
                                 float* __restrict__ deg) {
    const int i = blockIdx.x * blockDim.x + threadIdx.x;
    const int E = Ectx + Eint;
    if (i >= E) return;
    const int row = (i < Ectx) ? ctx_e[i] : int_e[i - Ectx];
    atomicAdd(&deg[row], 1.0f);
}

// ---------------------------------------------------------------------------
// node init: Xn, h = lin_in(feat), channel-attention softmax
// ---------------------------------------------------------------------------
__global__ void __launch_bounds__(128)
node_init_kernel(const float* __restrict__ Hn, const float* __restrict__ X,
                 const float* __restrict__ pos, const float* __restrict__ agt,
                 const float* __restrict__ aemb, const float* __restrict__ aw,
                 const float* __restrict__ beta,
                 const int* __restrict__ bid,
                 const float* __restrict__ csum, const float* __restrict__ ccnt,
                 const float* __restrict__ attw, const float* __restrict__ attb,
                 const float* __restrict__ liw, const float* __restrict__ lib,
                 float* __restrict__ Xn, float* __restrict__ xcur,
                 float* __restrict__ h, float* __restrict__ att)
{
    const int i = blockIdx.x;
    const int f = threadIdx.x;
    __shared__ float feat[279];
    __shared__ float logits[NCC];

    const float b = beta[i];
    feat[f] = Hn[i * HIDD + f];
    if (f == 0) { feat[128] = b; feat[129] = sinf(b); feat[130] = cosf(b); }
    if (f < 20) feat[131 + f] = agt[i * 20 + f];
    feat[151 + f] = pos[i * HIDD + f];
    if (f < NCC) {
        float acc = attb[0];
        for (int k = 0; k < 32; ++k) acc += aemb[(i * NCC + f) * 32 + k] * attw[k];
        logits[f] = (aw[i * NCC + f] > 0.0f) ? acc : -1e9f;
    }
    __syncthreads();

    float acc = lib[f];
    for (int k = 0; k < 279; ++k) acc += feat[k] * liw[k * HIDD + f];
    h[i * HIDD + f] = acc;

    if (f < NCC) {
        float mx = -1e30f;
        for (int k = 0; k < NCC; ++k) mx = fmaxf(mx, logits[k]);
        float den = 0.0f;
        for (int k = 0; k < NCC; ++k) den += __expf(logits[k] - mx);
        att[i * NCC + f] = __expf(logits[f] - mx) * __builtin_amdgcn_rcpf(den);
    }
    if (f < NCC * 3) {
        const int d = f % 3;
        const int bb = bid[i];
        const float cen = csum[bb * 3 + d] * __builtin_amdgcn_rcpf(ccnt[bb]);
        const float v = (X[i * NCC * 3 + f] - cen) * 0.1f;   // / STD(10)
        Xn[i * NCC * 3 + f] = v;
        xcur[i * NCC * 3 + f] = v;
    }
}

// ---------------------------------------------------------------------------
// node update: h += h2(silu(h1([h|agg]))), x += (xacc/deg)*atom_weights
// ---------------------------------------------------------------------------
__global__ void __launch_bounds__(128)
node_update_kernel(float* __restrict__ h, const float* __restrict__ agg,
                   float* __restrict__ xcur, const float* __restrict__ xacc,
                   const float* __restrict__ deg, const float* __restrict__ aw,
                   const float* __restrict__ h1w, const float* __restrict__ h1b,
                   const float* __restrict__ h2w, const float* __restrict__ h2b)
{
    const int i = blockIdx.x;
    const int f = threadIdx.x;
    __shared__ float cat[256];
    __shared__ float sh[HIDD];
    cat[f]       = h[i * HIDD + f];
    cat[128 + f] = agg[i * HIDD + f];
    __syncthreads();
    float a = h1b[f];
    for (int k = 0; k < 256; ++k) a += cat[k] * h1w[k * HIDD + f];
    sh[f] = fast_silu(a);
    if (f < NCC * 3) {
        const int c = f / 3;
        xcur[i * NCC * 3 + f] += xacc[i * NCC * 3 + f] *
            __builtin_amdgcn_rcpf(deg[i] + 1e-8f) * aw[i * NCC + c];
    }
    __syncthreads();
    float bsum = h2b[f];
    for (int k = 0; k < HIDD; ++k) bsum += sh[k] * h2w[k * HIDD + f];
    h[i * HIDD + f] = cat[f] + bsum;
}

// ---------------------------------------------------------------------------
// outputs
// ---------------------------------------------------------------------------
__global__ void eps_h_kernel(const float* __restrict__ h, const float* __restrict__ w,
                             const float* __restrict__ b, const float* __restrict__ Hn,
                             const unsigned char* __restrict__ mgen,
                             float* __restrict__ out, int N) {
    const int idx = blockIdx.x * blockDim.x + threadIdx.x;
    if (idx >= N * HIDD) return;
    const int i = idx >> 7, f = idx & 127;
    float acc = b[f];
    for (int k = 0; k < HIDD; ++k) acc += h[i * HIDD + k] * w[k * HIDD + f];
    out[idx] = (acc - Hn[idx]) * (mgen[i] ? 1.0f : 0.0f);
}

__global__ void eps_x_kernel(const float* __restrict__ xcur, const float* __restrict__ Xn,
                             const unsigned char* __restrict__ mgen,
                             float* __restrict__ out, int N) {
    const int idx = blockIdx.x * blockDim.x + threadIdx.x;
    if (idx >= N * NCC * 3) return;
    const int i = idx / (NCC * 3);
    out[idx] = (xcur[idx] - Xn[idx]) * (mgen[i] ? 1.0f : 0.0f);
}

// ---------------------------------------------------------------------------
// host-side launcher
// ---------------------------------------------------------------------------
extern "C" void kernel_launch(void* const* d_in, const int* in_sizes, int n_in,
                              void* d_out, int out_size, void* d_ws, size_t ws_size,
                              hipStream_t stream) {
    (void)n_in; (void)out_size; (void)ws_size;
    // --- input pointers (setup_inputs order; params flattened by sorted keys) ---
    const float* H_noisy   = (const float*)d_in[0];
    const float* X         = (const float*)d_in[1];
    const float* posemb    = (const float*)d_in[2];
    const float* atom_gt   = (const float*)d_in[3];
    const float* aemb      = (const float*)d_in[4];
    const float* aw        = (const float*)d_in[5];
    const float* beta      = (const float*)d_in[6];
    const float* att_b     = (const float*)d_in[7];
    const float* att_w     = (const float*)d_in[8];
    const float* edge_tab  = (const float*)d_in[9];
    struct LayerP { const float *e1b,*e1w,*e2b,*e2w,*h1b,*h1w,*h2b,*h2w,*rb,*rw,*xb,*xw; };
    LayerP lp[3];
    for (int l = 0; l < 3; ++l) {
        const int base = 10 + 12 * l;
        lp[l].e1b = (const float*)d_in[base+0];  lp[l].e1w = (const float*)d_in[base+1];
        lp[l].e2b = (const float*)d_in[base+2];  lp[l].e2w = (const float*)d_in[base+3];
        lp[l].h1b = (const float*)d_in[base+4];  lp[l].h1w = (const float*)d_in[base+5];
        lp[l].h2b = (const float*)d_in[base+6];  lp[l].h2w = (const float*)d_in[base+7];
        lp[l].rb  = (const float*)d_in[base+8];  lp[l].rw  = (const float*)d_in[base+9];
        lp[l].xb  = (const float*)d_in[base+10]; lp[l].xw  = (const float*)d_in[base+11];
    }
    const float* lin_in_b  = (const float*)d_in[46];
    const float* lin_in_w  = (const float*)d_in[47];
    const float* lin_out_b = (const float*)d_in[48];
    const float* lin_out_w = (const float*)d_in[49];
    const int* ctx_e = (const int*)d_in[50];
    const int* int_e = (const int*)d_in[51];
    const int* bid   = (const int*)d_in[52];
    const unsigned char* mgen  = (const unsigned char*)d_in[53];
    const unsigned char* amask = (const unsigned char*)d_in[54];

    const int N    = in_sizes[6];          // beta: one per node
    const int Ectx = in_sizes[50] / 2;
    const int Eint = in_sizes[51] / 2;
    const int E    = Ectx + Eint;

    // --- workspace layout ---
    char* wp = (char*)d_ws;
    auto take = [&](size_t bytes) -> char* {
        char* p = wp; wp += (bytes + 255) & ~(size_t)255; return p;
    };
    _Float16* Wr16[3];  _Float16* We116[3]; _Float16* We216[3];
    for (int l = 0; l < 3; ++l) {
        Wr16[l]  = (_Float16*)take((size_t)(KRAD / 32) * 4096 * 2);
        We116[l] = (_Float16*)take((size_t)(KE1  / 32) * 4096 * 2);
        We216[l] = (_Float16*)take((size_t)(KE2  / 32) * 4096 * 2);
    }
    float* Xn   = (float*)take((size_t)N * NCC * 3 * 4);
    float* xcur = (float*)take((size_t)N * NCC * 3 * 4);
    float* h    = (float*)take((size_t)N * HIDD * 4);
    float* att  = (float*)take((size_t)N * NCC * 4);
    float* csum = (float*)take((size_t)(64 * 3 + 64 + N) * 4);  // csum|ccnt|deg contiguous
    float* ccnt = csum + 64 * 3;
    float* deg  = ccnt + 64;
    float* agg  = (float*)take((size_t)(N * HIDD + N * NCC * 3) * 4); // agg|xacc contiguous
    float* xacc = agg + (size_t)N * HIDD;

    // --- 1. weight conversion + B-fragment swizzle (f16) ---
    for (int l = 0; l < 3; ++l) {
        swizzle_b_kernel<<<((KRAD / 32) * 4096 + 255) / 256, 256, 0, stream>>>(
            lp[l].rw, Wr16[l], NCC * NCC, KRAD / 32);
        swizzle_b_kernel<<<((KE1 / 32) * 4096 + 255) / 256, 256, 0, stream>>>(
            lp[l].e1w, We116[l], KE1, KE1 / 32);
        swizzle_b_kernel<<<((KE2 / 32) * 4096 + 255) / 256, 256, 0, stream>>>(
            lp[l].e2w, We216[l], KE2, KE2 / 32);
    }

    // --- 2. centers + degree ---
    const int nzero1 = 64 * 3 + 64 + N;
    zero_kernel<<<(nzero1 + 255) / 256, 256, 0, stream>>>(csum, nzero1);
    centers_accum_kernel<<<(N + 255) / 256, 256, 0, stream>>>(X, mgen, amask, bid, csum, ccnt, N);
    deg_accum_kernel<<<(E + 255) / 256, 256, 0, stream>>>(ctx_e, Ectx, int_e, Eint, deg);

    // --- 3. node init ---
    node_init_kernel<<<N, 128, 0, stream>>>(H_noisy, X, posemb, atom_gt, aemb, aw, beta,
                                            bid, csum, ccnt, att_w, att_b,
                                            lin_in_w, lin_in_b, Xn, xcur, h, att);

    // --- 4. message-passing layers ---
    const int nzero2 = N * HIDD + N * NCC * 3;
    const int eblocks = (E + 31) / 32;
    for (int l = 0; l < 3; ++l) {
        zero_kernel<<<(nzero2 + 255) / 256, 256, 0, stream>>>(agg, nzero2);
        edge_kernel<<<eblocks, 64, 0, stream>>>(xcur, att, h,
                                                ctx_e, Ectx, int_e, Eint, edge_tab,
                                                Wr16[l], lp[l].rb,
                                                We116[l], lp[l].e1b,
                                                We216[l], lp[l].e2b,
                                                lp[l].xw, lp[l].xb,
                                                agg, xacc);
        node_update_kernel<<<N, 128, 0, stream>>>(h, agg, xcur, xacc, deg, aw,
                                                  lp[l].h1w, lp[l].h1b,
                                                  lp[l].h2w, lp[l].h2b);
    }

    // --- 5. outputs: [eps_H (N*128) | eps_X (N*42)] ---
    float* out = (float*)d_out;
    eps_h_kernel<<<(N * HIDD + 255) / 256, 256, 0, stream>>>(h, lin_out_w, lin_out_b,
                                                             H_noisy, mgen, out, N);
    eps_x_kernel<<<(N * NCC * 3 + 255) / 256, 256, 0, stream>>>(xcur, Xn, mgen,
                                                                out + (size_t)N * HIDD, N);
}
